// RNNDecoder_73804718014861
// MI455X (gfx1250) — matrix operand order repaired and checked by reading, not compiled, not run on hardware
//
#include <hip/hip_runtime.h>
#include <hip/hip_bf16.h>

typedef __attribute__((ext_vector_type(16))) _Float16 v16h;
typedef __attribute__((ext_vector_type(8)))  float    v8f;

#define B_TOT   256
#define LAT     128
#define UPS     64
#define H       512
#define VOCAB   25
#define T_STEPS 500
#define BT      16      // batch rows per workgroup
#define NTHR    512     // 16 wave32 waves

// ---------------- WMMA helpers (CDNA5 16x16x32 f16 -> f32) ----------------

__device__ __forceinline__ v8f wmma16(v16h a, v16h b, v8f c) {
  // emits v_wmma_f32_16x16x32_f16
  return __builtin_amdgcn_wmma_f32_16x16x32_f16(false, a, false, b, (short)0, c,
                                                false, false);
}

// A-matrix fragment (16x32 f16) from an LDS row-major [16][kdim] f16 buffer.
// ISA layout: lane L<16: row M=L, K={0..7,16..23}; lane L>=16: row M=L-16,
// K={8..15,24..31}; VGPR v packs K pair ((v>=4?16:0)+half*8+(v&3)*2, +1).
__device__ __forceinline__ v16h load_a_lds(const _Float16* a16, int kdim, int kt,
                                           int lane) {
  const int row  = lane & 15;
  const int half = lane >> 4;
  const _Float16* base = a16 + row * kdim + kt * 32 + half * 8;
  v16h a;
  unsigned* ap = (unsigned*)&a;
#pragma unroll
  for (int v = 0; v < 8; ++v) {
    int kk = ((v >= 4) ? 16 : 0) + (v & 3) * 2;
    ap[v] = *(const unsigned*)(base + kk);
  }
  return a;
}

// B-matrix fragment (32x16 f16) from a pre-packed global buffer: fragment
// blocks stored contiguously as [nt][kt][lane][16 f16] -> two b128 loads.
__device__ __forceinline__ v16h load_b_pack(const _Float16* wp, int KT, int nt,
                                            int kt, int lane) {
  const uint4* p =
      (const uint4*)(wp + (((size_t)(nt * KT + kt)) * 32 + lane) * 16);
  v16h b;
  uint4* bp = (uint4*)&b;
  bp[0] = p[0];
  bp[1] = p[1];
  return b;
}

__device__ __forceinline__ float wred_max(float x) {
#pragma unroll
  for (int o = 16; o > 0; o >>= 1) x = fmaxf(x, __shfl_xor(x, o, 32));
  return x;
}
__device__ __forceinline__ float wred_sum(float x) {
#pragma unroll
  for (int o = 16; o > 0; o >>= 1) x += __shfl_xor(x, o, 32);
  return x;
}
__device__ __forceinline__ float sigmoidf_(float x) {
  return 1.0f / (1.0f + expf(-x));
}

// ---------------- Prep kernels (run once per launch, on stream) ----------------

// Pack a weight matrix into WMMA B-fragment layout.
// Logical B is [K][N] with B[k][n] = (mode==0) ? src[k*stride+n] : src[n*stride+k].
// ISA B layout per fragment (kt,nt): lane L: col N = nt*16+(L&15),
// K = kt*32 + (v>=4?16:0) + (L>>4)*8 + (v&3)*2 + j  (v = vgpr 0..7, j = f16 slot 0..1).
__global__ void pack_b_kernel(const float* __restrict__ src,
                              _Float16* __restrict__ dst, int K, int N,
                              int srcStride, int mode, int nValid) {
  const int KT = K >> 5;
  const size_t total = (size_t)K * N;
  for (size_t e = (size_t)blockIdx.x * blockDim.x + threadIdx.x; e < total;
       e += (size_t)gridDim.x * blockDim.x) {
    int e16  = (int)(e & 15);
    int v    = e16 >> 1;
    int j    = e16 & 1;
    int lane = (int)((e >> 4) & 31);
    size_t blk = e >> 9;
    int kt = (int)(blk % KT);
    int nt = (int)(blk / KT);
    int k = kt * 32 + ((v >= 4) ? 16 : 0) + (lane >> 4) * 8 + (v & 3) * 2 + j;
    int n = nt * 16 + (lane & 15);
    float val = 0.0f;
    if (n < nValid)
      val = (mode == 0) ? src[(size_t)k * srcStride + n]
                        : src[(size_t)n * srcStride + k];
    dst[e] = (_Float16)val;
  }
}

__global__ void cvt_f16_kernel(const float* __restrict__ src,
                               _Float16* __restrict__ dst, size_t n) {
  for (size_t i = (size_t)blockIdx.x * blockDim.x + threadIdx.x; i < n;
       i += (size_t)gridDim.x * blockDim.x)
    dst[i] = (_Float16)src[i];
}

// ---------------- Persistent decoder kernel ----------------
// One workgroup owns 16 batch rows for all 500 steps. 16 waves.

__global__ __launch_bounds__(NTHR, 1)
void decoder_kernel(const float* __restrict__ latent,
                    const float* __restrict__ embed,
                    const float* __restrict__ W_hid,
                    const float* __restrict__ b_hid,
                    const float* __restrict__ b_ih,
                    const float* __restrict__ b_hh,
                    const float* __restrict__ b_mem,
                    const float* __restrict__ b_cat,
                    const float* __restrict__ b_out,
                    const _Float16* __restrict__ up16,
                    const _Float16* __restrict__ Wih16,
                    const _Float16* __restrict__ Whh16,
                    const _Float16* __restrict__ WmemN16,
                    const _Float16* __restrict__ WmemT16,
                    const _Float16* __restrict__ Wcat16,
                    const _Float16* __restrict__ Wout16,
                    float* __restrict__ out) {
  __shared__ __align__(16) _Float16 sh_xcc[BT * H];        // x (embed) / cc(tanh) alias
  __shared__ __align__(16) _Float16 sh_cat[BT * 2 * H];    // [h16 | ctx16]
  __shared__ __align__(16) float    sh_h[BT * H];          // h master (f32)
  __shared__ __align__(16) float    sh_sc[BT * T_STEPS];   // scores / softmax w
  __shared__ __align__(16) float    sh_q[BT * UPS];        // attention query
  __shared__ __align__(16) _Float16 sh_s[BT * UPS];        // pooled up vector
  __shared__ __align__(16) float    sh_logits[BT * 32];
  __shared__ int sh_ids[BT];

  const int tid  = threadIdx.x;
  const int lane = tid & 31;
  const int wave = tid >> 5;
  const int base = blockIdx.x * BT;

  // ---- P0: init ids = SOS, h0 = latent @ W_hid^T + b_hid
  if (tid < BT) sh_ids[tid] = 0;  // SOS
  for (int idx = tid; idx < BT * H; idx += NTHR) {
    int r = idx >> 9;        // /H
    int c = idx & (H - 1);
    const float* lrow = latent + (size_t)(base + r) * LAT;
    const float* wrow = W_hid + (size_t)c * LAT;
    float acc = b_hid[c];
#pragma unroll 4
    for (int k = 0; k < LAT; ++k) acc += lrow[k] * wrow[k];
    sh_h[r * H + c] = acc;
    sh_cat[r * 2 * H + c] = (_Float16)acc;
  }
  __syncthreads();

  for (int t = 0; t < T_STEPS; ++t) {
    // ---- P1: embedding gather x = embed[ids]  (f16 into sh_xcc)
    for (int idx = tid; idx < BT * H; idx += NTHR) {
      int r = idx >> 9;
      int c = idx & (H - 1);
      sh_xcc[idx] = (_Float16)embed[(size_t)sh_ids[r] * H + c];
    }
    __syncthreads();

    // ---- P2: gate GEMMs. Per wave: 2 column tiles x {gi_r,gi_z,gi_n} then
    //          2 x {gh_r,gh_z,gh_n}. K-loops kept rolled to avoid VGPR spills;
    //          accumulators live in registers across the barrier.
    {
      v8f zf = {};
      v8f Ci[2][3], Ch[2][3];
#pragma unroll
      for (int it = 0; it < 2; ++it)
#pragma unroll
        for (int g = 0; g < 3; ++g) { Ci[it][g] = zf; Ch[it][g] = zf; }

#pragma unroll 1
      for (int kt = 0; kt < 16; ++kt) {
        v16h ax = load_a_lds(sh_xcc, H, kt, lane);
#pragma unroll
        for (int it = 0; it < 2; ++it) {
          int nt = wave * 2 + it;
          Ci[it][0] = wmma16(ax, load_b_pack(Wih16, 16, nt,      kt, lane), Ci[it][0]);
          Ci[it][1] = wmma16(ax, load_b_pack(Wih16, 16, 32 + nt, kt, lane), Ci[it][1]);
          Ci[it][2] = wmma16(ax, load_b_pack(Wih16, 16, 64 + nt, kt, lane), Ci[it][2]);
        }
      }
#pragma unroll 1
      for (int kt = 0; kt < 16; ++kt) {
        v16h ah = load_a_lds(sh_cat, 2 * H, kt, lane);  // h half of concat
#pragma unroll
        for (int it = 0; it < 2; ++it) {
          int nt = wave * 2 + it;
          Ch[it][0] = wmma16(ah, load_b_pack(Whh16, 16, nt,      kt, lane), Ch[it][0]);
          Ch[it][1] = wmma16(ah, load_b_pack(Whh16, 16, 32 + nt, kt, lane), Ch[it][1]);
          Ch[it][2] = wmma16(ah, load_b_pack(Whh16, 16, 64 + nt, kt, lane), Ch[it][2]);
        }
      }
      __syncthreads();  // all reads of sh_cat done before h_new is written

      // ---- P3: gate elementwise in registers, write h_new (f32 + f16)
      const int halfq = lane >> 4;
#pragma unroll
      for (int it = 0; it < 2; ++it) {
        int nt = wave * 2 + it;
        int jc = nt * 16 + (lane & 15);
        float bir = b_ih[jc], biz = b_ih[H + jc], bin_ = b_ih[2 * H + jc];
        float bhr = b_hh[jc], bhz = b_hh[H + jc], bhn  = b_hh[2 * H + jc];
#pragma unroll
        for (int v = 0; v < 8; ++v) {
          int m = v + 8 * halfq;
          float hold = sh_h[m * H + jc];
          float rg = sigmoidf_(Ci[it][0][v] + bir + Ch[it][0][v] + bhr);
          float zg = sigmoidf_(Ci[it][1][v] + biz + Ch[it][1][v] + bhz);
          float ng = tanhf(Ci[it][2][v] + bin_ + rg * (Ch[it][2][v] + bhn));
          float hn = (1.0f - zg) * ng + zg * hold;
          sh_h[m * H + jc] = hn;
          sh_cat[m * 2 * H + jc] = (_Float16)hn;
        }
      }
    }
    __syncthreads();

    // ---- P4: q = h_new @ W_mem   (M=16, N=64, K=512); waves 0..3
    if (wave < 4) {
      v8f c = {};
#pragma unroll 1
      for (int kt = 0; kt < 16; ++kt) {
        v16h a = load_a_lds(sh_cat, 2 * H, kt, lane);
        c = wmma16(a, load_b_pack(WmemN16, 16, wave, kt, lane), c);
      }
      int n = wave * 16 + (lane & 15);
      int halfq = lane >> 4;
#pragma unroll
      for (int v = 0; v < 8; ++v) sh_q[(v + 8 * halfq) * UPS + n] = c[v];
    }
    __syncthreads();

    // ---- P5: scores = q . up[b,t,:], softmax over T, s = up^T w.
    //          wave i <-> batch row i (mem never materialized; softmax shift-invariance
    //          kills the h.b_mem constant).
    {
      const int bi = base + wave;
      float qv[UPS];
#pragma unroll
      for (int u = 0; u < UPS; ++u) qv[u] = sh_q[wave * UPS + u];
      const _Float16* uprow = up16 + (size_t)bi * T_STEPS * UPS;

      float mx = -3.4e38f;
#pragma unroll 1
      for (int t2 = lane; t2 < T_STEPS; t2 += 32) {
        const uint4* p = (const uint4*)(uprow + (size_t)t2 * UPS);
        float acc = 0.0f;
#pragma unroll
        for (int c8 = 0; c8 < 8; ++c8) {
          uint4 raw = p[c8];
          const _Float16* hp = (const _Float16*)&raw;
#pragma unroll
          for (int e = 0; e < 8; ++e) acc += qv[c8 * 8 + e] * (float)hp[e];
        }
        sh_sc[wave * T_STEPS + t2] = acc;
        mx = fmaxf(mx, acc);
      }
      mx = wred_max(mx);

      float sum = 0.0f;
#pragma unroll 1
      for (int t2 = lane; t2 < T_STEPS; t2 += 32) {
        float e = expf(sh_sc[wave * T_STEPS + t2] - mx);
        sh_sc[wave * T_STEPS + t2] = e;
        sum += e;
      }
      sum = wred_sum(sum);
      float inv = 1.0f / sum;

      float su[UPS];
#pragma unroll
      for (int u = 0; u < UPS; ++u) su[u] = 0.0f;
#pragma unroll 1
      for (int t2 = lane; t2 < T_STEPS; t2 += 32) {
        float w = sh_sc[wave * T_STEPS + t2] * inv;
        const uint4* p = (const uint4*)(uprow + (size_t)t2 * UPS);
#pragma unroll
        for (int c8 = 0; c8 < 8; ++c8) {
          uint4 raw = p[c8];
          const _Float16* hp = (const _Float16*)&raw;
#pragma unroll
          for (int e = 0; e < 8; ++e) su[c8 * 8 + e] += w * (float)hp[e];
        }
      }
#pragma unroll
      for (int u = 0; u < UPS; ++u) {
        float tot = wred_sum(su[u]);
        if (lane == 0) sh_s[wave * UPS + u] = (_Float16)tot;
      }
    }
    __syncthreads();

    // ---- P6: ctx = s @ W_mem^T + b_mem  (M=16, N=512, K=64) -> concat high half
    {
      v8f c0 = {}, c1 = {};
#pragma unroll
      for (int kt = 0; kt < 2; ++kt) {
        v16h a = load_a_lds(sh_s, UPS, kt, lane);
        c0 = wmma16(a, load_b_pack(WmemT16, 2, wave * 2,     kt, lane), c0);
        c1 = wmma16(a, load_b_pack(WmemT16, 2, wave * 2 + 1, kt, lane), c1);
      }
      int halfq = lane >> 4;
#pragma unroll
      for (int it = 0; it < 2; ++it) {
        int n = (wave * 2 + it) * 16 + (lane & 15);
        float bm = b_mem[n];
        const v8f& c = it ? c1 : c0;
#pragma unroll
        for (int v = 0; v < 8; ++v)
          sh_cat[(v + 8 * halfq) * 2 * H + H + n] = (_Float16)(c[v] + bm);
      }
    }
    __syncthreads();

    // ---- P7: cc = tanh(concat @ W_cat^T + b_cat)  (M=16, N=512, K=1024)
    {
      v8f c0 = {}, c1 = {};
#pragma unroll 1
      for (int kt = 0; kt < 32; ++kt) {
        v16h a = load_a_lds(sh_cat, 2 * H, kt, lane);
        c0 = wmma16(a, load_b_pack(Wcat16, 32, wave * 2,     kt, lane), c0);
        c1 = wmma16(a, load_b_pack(Wcat16, 32, wave * 2 + 1, kt, lane), c1);
      }
      int halfq = lane >> 4;
#pragma unroll
      for (int it = 0; it < 2; ++it) {
        int n = (wave * 2 + it) * 16 + (lane & 15);
        float bc = b_cat[n];
        const v8f& c = it ? c1 : c0;
#pragma unroll
        for (int v = 0; v < 8; ++v)
          sh_xcc[(v + 8 * halfq) * H + n] = (_Float16)tanhf(c[v] + bc);
      }
    }
    __syncthreads();

    // ---- P8: logits = cc @ W_out^T + b_out  (M=16, N=32 padded, K=512); waves 0..1
    if (wave < 2) {
      v8f c = {};
#pragma unroll 1
      for (int kt = 0; kt < 16; ++kt) {
        v16h a = load_a_lds(sh_xcc, H, kt, lane);
        c = wmma16(a, load_b_pack(Wout16, 16, wave, kt, lane), c);
      }
      int n = wave * 16 + (lane & 15);
      float bo = (n < VOCAB) ? b_out[n] : 0.0f;
      int halfq = lane >> 4;
#pragma unroll
      for (int v = 0; v < 8; ++v) sh_logits[(v + 8 * halfq) * 32 + n] = c[v] + bo;
    }
    __syncthreads();

    // ---- P9: emit logits to out[B,V,T]; greedy argmax (first-max) feedback
    for (int idx = tid; idx < BT * VOCAB; idx += NTHR) {
      int r = idx / VOCAB;
      int vv = idx - r * VOCAB;
      out[((size_t)(base + r) * VOCAB + vv) * T_STEPS + t] = sh_logits[r * 32 + vv];
    }
    if (tid < BT) {
      const float* lg = sh_logits + tid * 32;
      float best = lg[0];
      int bidx = 0;
#pragma unroll
      for (int vv = 1; vv < VOCAB; ++vv) {
        float x = lg[vv];
        if (x > best) { best = x; bidx = vv; }
      }
      sh_ids[tid] = bidx;
    }
    __syncthreads();
  }
}

// ---------------- Host launch ----------------

extern "C" void kernel_launch(void* const* d_in, const int* in_sizes, int n_in,
                              void* d_out, int out_size, void* d_ws, size_t ws_size,
                              hipStream_t stream) {
  (void)in_sizes; (void)n_in; (void)out_size; (void)ws_size;
  const float* latent = (const float*)d_in[0];
  const float* up     = (const float*)d_in[1];
  const float* embed  = (const float*)d_in[2];
  const float* W_hid  = (const float*)d_in[3];
  const float* b_hid  = (const float*)d_in[4];
  const float* W_ih   = (const float*)d_in[5];
  const float* b_ih   = (const float*)d_in[6];
  const float* W_hh   = (const float*)d_in[7];
  const float* b_hh   = (const float*)d_in[8];
  const float* W_mem  = (const float*)d_in[9];
  const float* b_mem  = (const float*)d_in[10];
  const float* W_cat  = (const float*)d_in[11];
  const float* b_cat  = (const float*)d_in[12];
  const float* W_out  = (const float*)d_in[13];
  const float* b_out  = (const float*)d_in[14];
  float* out = (float*)d_out;

  char* ws = (char*)d_ws;
  size_t off = 0;
  _Float16* up16    = (_Float16*)(ws + off); off += (size_t)B_TOT * T_STEPS * UPS * 2;
  _Float16* Wih16   = (_Float16*)(ws + off); off += (size_t)3 * H * H * 2;
  _Float16* Whh16   = (_Float16*)(ws + off); off += (size_t)3 * H * H * 2;
  _Float16* WmemN16 = (_Float16*)(ws + off); off += (size_t)H * UPS * 2;
  _Float16* WmemT16 = (_Float16*)(ws + off); off += (size_t)UPS * H * 2;
  _Float16* Wcat16  = (_Float16*)(ws + off); off += (size_t)2 * H * H * 2;
  _Float16* Wout16  = (_Float16*)(ws + off); off += (size_t)H * 32 * 2;

  // Prep: f16 copy of up, weights pre-packed into WMMA B-fragment layout.
  cvt_f16_kernel<<<2048, 256, 0, stream>>>(up, up16, (size_t)B_TOT * T_STEPS * UPS);
  pack_b_kernel<<<1024, 256, 0, stream>>>(W_ih, Wih16, H, 3 * H, H, 1, 3 * H);
  pack_b_kernel<<<1024, 256, 0, stream>>>(W_hh, Whh16, H, 3 * H, H, 1, 3 * H);
  pack_b_kernel<<<256,  256, 0, stream>>>(W_mem, WmemN16, H, UPS, UPS, 0, UPS);
  pack_b_kernel<<<256,  256, 0, stream>>>(W_mem, WmemT16, UPS, H, UPS, 1, H);
  pack_b_kernel<<<1024, 256, 0, stream>>>(W_cat, Wcat16, 2 * H, H, 2 * H, 1, H);
  pack_b_kernel<<<128,  256, 0, stream>>>(W_out, Wout16, H, 32, H, 1, VOCAB);

  // Persistent decoder: 16 WGs x 16 waves, full 500-step recurrence on-chip.
  decoder_kernel<<<B_TOT / BT, NTHR, 0, stream>>>(
      latent, embed, W_hid, b_hid, b_ih, b_hh, b_mem, b_cat, b_out,
      up16, Wih16, Whh16, WmemN16, WmemT16, Wcat16, Wout16, out);
}
